// HOPE_13898514170298
// MI455X (gfx1250) — matrix operand order repaired and checked by reading, not compiled, not run on hardware
//
#include <hip/hip_runtime.h>
#include <hip/hip_bf16.h>
#include <math.h>

#define S_LEN 2048
#define DMODEL 512
#define NHEADS 8
#define HDIM 64
#define NCHUNK 32
#define CHUNK 64
#define HIDDEN_DIM 2048
#define NLAYERS 4
#define NLEVELS 3
#define VOCAB_N 50257

typedef float v8f __attribute__((ext_vector_type(8)));
typedef __bf16 v16bf __attribute__((ext_vector_type(16)));
typedef unsigned int u32x4 __attribute__((ext_vector_type(4)));
typedef float f32x4 __attribute__((ext_vector_type(4)));

__device__ __forceinline__ unsigned int pack_bf16x2(float x, float y) {
  unsigned int ux = __float_as_uint(x);
  unsigned int uy = __float_as_uint(y);
  ux += 0x7fffu + ((ux >> 16) & 1u);
  uy += 0x7fffu + ((uy >> 16) & 1u);
  return (ux >> 16) | (uy & 0xffff0000u);
}
__device__ __forceinline__ unsigned short f2bf(float x) {
  unsigned int u = __float_as_uint(x);
  u += 0x7fffu + ((u >> 16) & 1u);
  return (unsigned short)(u >> 16);
}
__device__ __forceinline__ float gelu_f(float x) {
  float x3 = x * x * x;
  return 0.5f * x * (1.f + tanhf(0.7978845608028654f * (x + 0.044715f * x3)));
}
__device__ __forceinline__ float elu1_f(float x) {
  return x > 0.f ? x + 1.f : __expf(x);
}

// C[M,N] = act(A[M,K] @ B + bias) + res.  B is K x N (BT=false) or N x K (BT=true).
// ACT: 0 = none, 1 = elu(x)+1, 2 = gelu(tanh approx). M multiple of 128; K multiple of 32.
template <int ACT, bool BT>
__global__ __launch_bounds__(256) void gemm_wmma_kernel(
    const float* __restrict__ A, const float* __restrict__ B,
    const float* __restrict__ bias, const float* __restrict__ res,
    float* __restrict__ C, int M, int N, int K) {
  (void)M;
  __shared__ unsigned short As[128 * 32];
  __shared__ unsigned short Bs[128 * 32];  // stored as [n][k]

  const int tid = threadIdx.x;
  const int lane = tid & 31;
  const int wid = tid >> 5;
  const int wm = wid & 3;   // wave m index (0..3) -> 32 rows each
  const int wn = wid >> 2;  // wave n index (0..1) -> 64 cols each
  const int bm = blockIdx.y * 128;
  const int bn = blockIdx.x * 128;
  const int hh = lane >> 4;   // lane half
  const int l16 = lane & 15;

  v8f acc[2][4];
#pragma unroll
  for (int i = 0; i < 2; ++i)
#pragma unroll
    for (int j = 0; j < 4; ++j)
#pragma unroll
      for (int r = 0; r < 8; ++r) acc[i][j][r] = 0.f;

  for (int k0 = 0; k0 < K; k0 += 32) {
    // ---- stage A tile (128 x 32) as bf16, row-major [r][k]
    {
      const int r = tid >> 1;
      const int c0 = (tid & 1) * 16;
      const f32x4* src = (const f32x4*)(A + (size_t)(bm + r) * K + k0 + c0);
      unsigned int* dst = (unsigned int*)&As[r * 32 + c0];
#pragma unroll
      for (int ii = 0; ii < 4; ++ii) {
        f32x4 t = src[ii];
        dst[2 * ii + 0] = pack_bf16x2(t[0], t[1]);
        dst[2 * ii + 1] = pack_bf16x2(t[2], t[3]);
      }
    }
    // ---- stage B tile as bf16, transposed layout [n][k]
    if (BT) {
      const int r = tid >> 1;
      const int c0 = (tid & 1) * 16;
      const int n = bn + r;
      unsigned int* dst = (unsigned int*)&Bs[r * 32 + c0];
      if (n < N) {
        const f32x4* src = (const f32x4*)(B + (size_t)n * K + k0 + c0);
#pragma unroll
        for (int ii = 0; ii < 4; ++ii) {
          f32x4 t = src[ii];
          dst[2 * ii + 0] = pack_bf16x2(t[0], t[1]);
          dst[2 * ii + 1] = pack_bf16x2(t[2], t[3]);
        }
      } else {
#pragma unroll
        for (int ii = 0; ii < 8; ++ii) dst[ii] = 0u;
      }
    } else {
      const int kk = tid >> 3;         // 0..31
      const int n0 = (tid & 7) * 16;   // 0..112
      const f32x4* src = (const f32x4*)(B + (size_t)(k0 + kk) * N + bn + n0);
#pragma unroll
      for (int ii = 0; ii < 4; ++ii) {
        f32x4 t = src[ii];
#pragma unroll
        for (int jj = 0; jj < 4; ++jj)
          Bs[(n0 + ii * 4 + jj) * 32 + kk] = f2bf(t[jj]);
      }
    }
    __syncthreads();

    union FragU { u32x4 u[2]; v16bf v; };
    v16bf afrag[2], bfrag[4];
    // A fragment: lane row = l16, element e -> K = e + hh*8 + (e>=8 ? 8 : 0)
#pragma unroll
    for (int mf = 0; mf < 2; ++mf) {
      const int row = wm * 32 + mf * 16 + l16;
      const u32x4* p = (const u32x4*)&As[row * 32];
      FragU f;
      f.u[0] = p[hh];
      f.u[1] = p[2 + hh];
      afrag[mf] = f.v;
    }
    // B fragment: lane col = l16, element e -> K = e + hh*8 + (e>=8 ? 8 : 0)
#pragma unroll
    for (int nf = 0; nf < 4; ++nf) {
      const int col = wn * 64 + nf * 16 + l16;
      const u32x4* p = (const u32x4*)&Bs[col * 32];
      FragU f;
      f.u[0] = p[hh];
      f.u[1] = p[2 + hh];
      bfrag[nf] = f.v;
    }
#pragma unroll
    for (int mf = 0; mf < 2; ++mf)
#pragma unroll
      for (int nf = 0; nf < 4; ++nf)
        acc[mf][nf] = __builtin_amdgcn_wmma_f32_16x16x32_bf16(
            false, afrag[mf], false, bfrag[nf], (short)0, acc[mf][nf], false, false);
    __syncthreads();
  }

  // ---- epilogue: D element r -> row = base + hh*8 + r, col = base + l16
#pragma unroll
  for (int mf = 0; mf < 2; ++mf) {
#pragma unroll
    for (int nf = 0; nf < 4; ++nf) {
      const int col = bn + wn * 64 + nf * 16 + l16;
      if (col < N) {
        const int row0 = bm + wm * 32 + mf * 16 + hh * 8;
        const float bv = bias ? bias[col] : 0.f;
#pragma unroll
        for (int r = 0; r < 8; ++r) {
          float v = acc[mf][nf][r] + bv;
          if (ACT == 1) v = elu1_f(v);
          if (ACT == 2) v = gelu_f(v);
          const size_t idx = (size_t)(row0 + r) * N + col;
          if (res) v += res[idx];
          C[idx] = v;
        }
      }
    }
  }
}

// LayerNorm over D=512, one block (256 threads) per row.
__global__ __launch_bounds__(256) void layernorm_kernel(
    const float* __restrict__ x, const float* __restrict__ w,
    const float* __restrict__ b, float* __restrict__ out) {
  const int row = blockIdx.x;
  const int tid = threadIdx.x;
  const float* xr = x + (size_t)row * DMODEL;
  float a0 = xr[tid];
  float a1 = xr[tid + 256];
  __shared__ float red[256];
  red[tid] = a0 + a1;
  __syncthreads();
  for (int s = 128; s > 0; s >>= 1) {
    if (tid < s) red[tid] += red[tid + s];
    __syncthreads();
  }
  const float mean = red[0] * (1.f / DMODEL);
  __syncthreads();
  const float d0 = a0 - mean, d1 = a1 - mean;
  red[tid] = d0 * d0 + d1 * d1;
  __syncthreads();
  for (int s = 128; s > 0; s >>= 1) {
    if (tid < s) red[tid] += red[tid + s];
    __syncthreads();
  }
  const float rstd = rsqrtf(red[0] * (1.f / DMODEL) + 1e-5f);
  float* orow = out + (size_t)row * DMODEL;
  orow[tid] = d0 * rstd * w[tid] + b[tid];
  orow[tid + 256] = d1 * rstd * w[tid + 256] + b[tid + 256];
}

__global__ void embed_kernel(const int* __restrict__ ids,
                             const float* __restrict__ tok,
                             const float* __restrict__ pos,
                             float* __restrict__ x) {
  const int idx = blockIdx.x * blockDim.x + threadIdx.x;
  const int s = idx >> 9;
  const int d = idx & 511;
  x[idx] = tok[(size_t)ids[s] * DMODEL + d] + pos[idx];
}

// Phase A: per (chunk, head): S_c = K_c^T V_c (64x64) and ksum_c = sum_t k_t.
__global__ __launch_bounds__(64) void attn_chunk_sum_kernel(
    const float* __restrict__ k, const float* __restrict__ v,
    float* __restrict__ states, float* __restrict__ ksums) {
  const int c = blockIdx.x;
  const int h = blockIdx.y;
  const int j = threadIdx.x;
  __shared__ float kS[CHUNK * HDIM];
  __shared__ float vS[CHUNK * HDIM];
  for (int t = 0; t < CHUNK; ++t) {
    const size_t g = (size_t)(c * CHUNK + t) * DMODEL + h * HDIM + j;
    kS[t * HDIM + j] = k[g];
    vS[t * HDIM + j] = v[g];
  }
  __syncthreads();
  float ks = 0.f;
#pragma unroll 8
  for (int t = 0; t < CHUNK; ++t) ks += kS[t * HDIM + j];
  ksums[((size_t)h * NCHUNK + c) * HDIM + j] = ks;
  const size_t base = ((size_t)h * NCHUNK + c) * HDIM * HDIM;
  for (int i = 0; i < HDIM; ++i) {
    float acc = 0.f;
#pragma unroll 8
    for (int t = 0; t < CHUNK; ++t) acc += kS[t * HDIM + i] * vS[t * HDIM + j];
    states[base + i * HDIM + j] = acc;
  }
}

// Phase B: exclusive prefix over chunks (per head). Thread j owns column j of M.
__global__ __launch_bounds__(64) void attn_prefix_kernel(
    float* __restrict__ states, float* __restrict__ ksums) {
  const int h = blockIdx.x;
  const int j = threadIdx.x;
  float run[HDIM];
#pragma unroll
  for (int i = 0; i < HDIM; ++i) run[i] = 0.f;
  float runk = 0.f;
  for (int c = 0; c < NCHUNK; ++c) {
    const size_t base = ((size_t)h * NCHUNK + c) * HDIM * HDIM;
#pragma unroll
    for (int i = 0; i < HDIM; ++i) {
      const float t = states[base + i * HDIM + j];
      states[base + i * HDIM + j] = run[i];
      run[i] += t;
    }
    const size_t kb = ((size_t)h * NCHUNK + c) * HDIM + j;
    const float tk = ksums[kb];
    ksums[kb] = runk;
    runk += tk;
  }
}

// Phase C: per (chunk, head): inclusive intra-chunk recurrence on top of chunk-start state.
__global__ __launch_bounds__(64) void attn_output_kernel(
    const float* __restrict__ q, const float* __restrict__ k,
    const float* __restrict__ v, const float* __restrict__ states,
    const float* __restrict__ ksums, float* __restrict__ y) {
  const int c = blockIdx.x;
  const int h = blockIdx.y;
  const int j = threadIdx.x;
  __shared__ float kS[CHUNK * HDIM];
  __shared__ float vS[CHUNK * HDIM];
  __shared__ float qS[CHUNK * HDIM];
  __shared__ float kc[CHUNK * HDIM];
  for (int t = 0; t < CHUNK; ++t) {
    const size_t g = (size_t)(c * CHUNK + t) * DMODEL + h * HDIM + j;
    kS[t * HDIM + j] = k[g];
    vS[t * HDIM + j] = v[g];
    qS[t * HDIM + j] = q[g];
  }
  __syncthreads();
  {
    float run = ksums[((size_t)h * NCHUNK + c) * HDIM + j];  // exclusive prefix
    for (int t = 0; t < CHUNK; ++t) {
      run += kS[t * HDIM + j];  // inclusive of k_t (matches reference order)
      kc[t * HDIM + j] = run;
    }
  }
  float Mcol[HDIM];
  const size_t base = ((size_t)h * NCHUNK + c) * HDIM * HDIM;
#pragma unroll
  for (int i = 0; i < HDIM; ++i) Mcol[i] = states[base + i * HDIM + j];
  __syncthreads();
  for (int t = 0; t < CHUNK; ++t) {
    const float vj = vS[t * HDIM + j];
    float num = 0.f, den = 0.f;
#pragma unroll
    for (int i = 0; i < HDIM; ++i) {
      const float qi = qS[t * HDIM + i];
      const float mi = Mcol[i] + kS[t * HDIM + i] * vj;  // M updated before use
      Mcol[i] = mi;
      num += mi * qi;
      den += kc[t * HDIM + i] * qi;
    }
    y[(size_t)(c * CHUNK + t) * DMODEL + h * HDIM + j] = num / (den + 1e-6f);
  }
}

extern "C" void kernel_launch(void* const* d_in, const int* in_sizes, int n_in,
                              void* d_out, int out_size, void* d_ws, size_t ws_size,
                              hipStream_t stream) {
  (void)in_sizes; (void)n_in; (void)out_size; (void)ws_size;
  const float* tok_emb = (const float*)d_in[0];
  const float* pos_emb = (const float*)d_in[1];
  const float* ln1_w = (const float*)d_in[2];
  const float* ln1_b = (const float*)d_in[3];
  const float* wq = (const float*)d_in[4];
  const float* wk = (const float*)d_in[5];
  const float* wv = (const float*)d_in[6];
  const float* wo = (const float*)d_in[7];
  const float* ln2_w = (const float*)d_in[8];
  const float* ln2_b = (const float*)d_in[9];
  const float* cms_w1 = (const float*)d_in[10];
  const float* cms_b1 = (const float*)d_in[11];
  const float* cms_w2 = (const float*)d_in[12];
  const float* cms_b2 = (const float*)d_in[13];
  const float* lnf_w = (const float*)d_in[14];
  const float* lnf_b = (const float*)d_in[15];
  const int* ids = (const int*)d_in[16];
  float* out = (float*)d_out;

  float* ws = (float*)d_ws;
  const size_t SD = (size_t)S_LEN * DMODEL;  // 1,048,576 floats
  float* x = ws;
  float* xln = x + SD;
  float* qb = xln + SD;
  float* kb = qb + SD;
  float* vb = kb + SD;
  float* yb = vb + SD;
  float* st = yb + SD;          // H*NC*64*64 == SD floats
  float* ksum = st + SD;        // H*NC*64 = 16384 floats
  float* h1 = ksum + 16384;     // S*HIDDEN = 4*SD floats
  float* h2 = h1 + (size_t)S_LEN * HIDDEN_DIM;

  const dim3 blk(256);
  embed_kernel<<<dim3((unsigned)(SD / 256)), blk, 0, stream>>>(ids, tok_emb, pos_emb, x);

  const dim3 g512(DMODEL / 128, S_LEN / 128);
  const dim3 gHID(HIDDEN_DIM / 128, S_LEN / 128);

  for (int l = 0; l < NLAYERS; ++l) {
    layernorm_kernel<<<S_LEN, blk, 0, stream>>>(x, ln1_w + l * DMODEL, ln1_b + l * DMODEL, xln);
    const size_t wo_off = (size_t)l * DMODEL * DMODEL;
    gemm_wmma_kernel<1, false><<<g512, blk, 0, stream>>>(
        xln, wq + wo_off, nullptr, nullptr, qb, S_LEN, DMODEL, DMODEL);
    gemm_wmma_kernel<1, false><<<g512, blk, 0, stream>>>(
        xln, wk + wo_off, nullptr, nullptr, kb, S_LEN, DMODEL, DMODEL);
    gemm_wmma_kernel<0, false><<<g512, blk, 0, stream>>>(
        xln, wv + wo_off, nullptr, nullptr, vb, S_LEN, DMODEL, DMODEL);

    attn_chunk_sum_kernel<<<dim3(NCHUNK, NHEADS), dim3(64), 0, stream>>>(kb, vb, st, ksum);
    attn_prefix_kernel<<<NHEADS, dim3(64), 0, stream>>>(st, ksum);
    attn_output_kernel<<<dim3(NCHUNK, NHEADS), dim3(64), 0, stream>>>(qb, kb, vb, st, ksum, yb);

    gemm_wmma_kernel<0, false><<<g512, blk, 0, stream>>>(
        yb, wo + wo_off, nullptr, x, x, S_LEN, DMODEL, DMODEL);

    layernorm_kernel<<<S_LEN, blk, 0, stream>>>(x, ln2_w + l * DMODEL, ln2_b + l * DMODEL, xln);
    const float* hin = xln;
    for (int lvl = 0; lvl < NLEVELS; ++lvl) {
      const size_t wi = (size_t)(l * NLEVELS + lvl);
      gemm_wmma_kernel<2, false><<<gHID, blk, 0, stream>>>(
          hin, cms_w1 + wi * DMODEL * HIDDEN_DIM, cms_b1 + wi * HIDDEN_DIM, nullptr,
          h1, S_LEN, HIDDEN_DIM, DMODEL);
      const bool last = (lvl == NLEVELS - 1);
      gemm_wmma_kernel<0, false><<<g512, blk, 0, stream>>>(
          h1, cms_w2 + wi * HIDDEN_DIM * DMODEL, cms_b2 + wi * DMODEL,
          last ? x : nullptr, last ? x : h2, S_LEN, DMODEL, HIDDEN_DIM);
      hin = h2;
    }
  }

  layernorm_kernel<<<S_LEN, blk, 0, stream>>>(x, lnf_w, lnf_b, xln);
  const dim3 gV((VOCAB_N + 127) / 128, S_LEN / 128);
  gemm_wmma_kernel<0, true><<<gV, blk, 0, stream>>>(
      xln, tok_emb, nullptr, nullptr, out, S_LEN, VOCAB_N, DMODEL);
}
